// PU_plus_NTXent_61332132987375
// MI455X (gfx1250) — matrix operand order repaired and checked by reading, not compiled, not run on hardware
//
#include <hip/hip_runtime.h>
#include <hip/hip_bf16.h>

typedef __attribute__((ext_vector_type(16))) _Float16 v16h;
typedef __attribute__((ext_vector_type(8)))  _Float16 v8h;
typedef __attribute__((ext_vector_type(8)))  float    v8f;
typedef __attribute__((ext_vector_type(2)))  _Float16 h2;
typedef __attribute__((ext_vector_type(4)))  _Float16 v4h;
typedef __attribute__((__vector_size__(16))) int      vi4;   // matches builtin param

#define BATCH   4096
#define LATENT  2048
#define ZDIM    128
#define NTOT    8192            // 2*BATCH
#define INV_TEMP 2.0f           // 1/0.5
#define SIM_MAX  2.0f           // |cos| <= 1  ->  sim <= 2 : fixed logsumexp shift
#define PRIOR      0.3f
#define PRIOR_P    0.5f
#define BPAD     136            // padded LDS row stride (halfs): 272B -> 4-bank skew
#define CHUNK    64             // columns staged per LDS chunk (4 WMMA tiles)

#define AS1 __attribute__((address_space(1)))
#define AS3 __attribute__((address_space(3)))

#ifdef __has_builtin
#  if __has_builtin(__builtin_amdgcn_global_load_async_to_lds_b128)
#    define ASYNC_LDS_COPY 1
#  endif
#endif
#ifndef ASYNC_LDS_COPY
#  define ASYNC_LDS_COPY 0
#endif

// ---------------------------------------------------------------------------
// Kernel 1: row-normalize z = [z_i ; z_j] -> f16, zn[8192][128]
// One wave per row: 32 lanes x float4 = 128 elements exactly.
// ---------------------------------------------------------------------------
__global__ void __launch_bounds__(256) normalize_z_kernel(
    const float* __restrict__ z_i, const float* __restrict__ z_j,
    _Float16* __restrict__ znh) {
  const int lane = threadIdx.x & 31;
  const int wave = threadIdx.x >> 5;
  const int row  = blockIdx.x * 8 + wave;              // grid = 1024 -> 8192 rows
  const float* src = (row < BATCH) ? (z_i + (size_t)row * ZDIM)
                                   : (z_j + (size_t)(row - BATCH) * ZDIM);
  float4 v = ((const float4*)src)[lane];
  float ss = v.x * v.x + v.y * v.y + v.z * v.z + v.w * v.w;
  #pragma unroll
  for (int off = 16; off > 0; off >>= 1) ss += __shfl_xor(ss, off, 32);
  float inv = 1.0f / fmaxf(sqrtf(ss), 1e-8f);
  v4h o;
  o.x = (_Float16)(v.x * inv);
  o.y = (_Float16)(v.y * inv);
  o.z = (_Float16)(v.z * inv);
  o.w = (_Float16)(v.w * inv);
  *(v4h*)(znh + (size_t)row * ZDIM + lane * 4) = o;
}

// ---------------------------------------------------------------------------
// Kernel 2: fused GEMV logits + PU-loss partial sums.
// One wave per row; computes logit for h_i and h_j sharing the W loads.
// Per-block output (8 floats): {cnt_pos, cnt_unl, ypos_i, yposinv_i, yunl_i,
//                               ypos_j, yposinv_j, yunl_j}
// ---------------------------------------------------------------------------
__device__ __forceinline__ float sigm(float x) { return 1.0f / (1.0f + __expf(-x)); }

__global__ void __launch_bounds__(256) gemv_pu_kernel(
    const float* __restrict__ h_i, const float* __restrict__ h_j,
    const float* __restrict__ W, const float* __restrict__ bptr,
    const int* __restrict__ target, float* __restrict__ pu_part) {
  const int lane = threadIdx.x & 31;
  const int wave = threadIdx.x >> 5;
  const int row  = blockIdx.x * 8 + wave;              // grid = 512 -> 4096 rows
  const float4* hi4 = (const float4*)(h_i + (size_t)row * LATENT);
  const float4* hj4 = (const float4*)(h_j + (size_t)row * LATENT);
  const float4* W4  = (const float4*)W;
  float di = 0.f, dj = 0.f;
  for (int t = lane; t < LATENT / 4; t += 32) {
    float4 w = W4[t];
    float4 a = hi4[t];
    float4 c = hj4[t];
    di += a.x * w.x + a.y * w.y + a.z * w.z + a.w * w.w;
    dj += c.x * w.x + c.y * w.y + c.z * w.z + c.w * w.w;
  }
  #pragma unroll
  for (int off = 16; off > 0; off >>= 1) {
    di += __shfl_xor(di, off, 32);
    dj += __shfl_xor(dj, off, 32);
  }
  __shared__ float part[8][8];
  if (lane == 0) {
    float bb = bptr[0];
    float li = di + bb, lj = dj + bb;
    float isPos = (target[row] == 1) ? 1.0f : 0.0f;
    float isUnl = 1.0f - isPos;
    part[wave][0] = isPos;
    part[wave][1] = isUnl;
    part[wave][2] = isPos * sigm(-li);   // y_pos
    part[wave][3] = isPos * sigm(li);    // y_pos_inv
    part[wave][4] = isUnl * sigm(li);    // y_unl
    part[wave][5] = isPos * sigm(-lj);
    part[wave][6] = isPos * sigm(lj);
    part[wave][7] = isUnl * sigm(lj);
  }
  __syncthreads();
  if (threadIdx.x < 8) {
    float s = 0.f;
    #pragma unroll
    for (int w2 = 0; w2 < 8; ++w2) s += part[w2][threadIdx.x];
    pu_part[blockIdx.x * 8 + threadIdx.x] = s;
  }
}

// ---------------------------------------------------------------------------
// Global -> LDS chunk copy: CHUNK(64) zn rows x 128 halfs, padded row stride.
// Async (ASYNCcnt-tracked gfx1250 path) when available, else load + ds_store.
// 128 threads x 8 x b128 each.
// ---------------------------------------------------------------------------
__device__ __forceinline__ void stage_chunk(const _Float16* __restrict__ g,
                                            _Float16* l, int tid) {
  #pragma unroll
  for (int q = 0; q < 8; ++q) {
    int idx = q * 128 + tid;
    int row = idx >> 4, seg = idx & 15;
    const _Float16* gp = g + (size_t)row * ZDIM + seg * 8;
    _Float16* lp = l + row * BPAD + seg * 8;
#if ASYNC_LDS_COPY
    __builtin_amdgcn_global_load_async_to_lds_b128(
        (AS1 vi4*)gp, (AS3 vi4*)lp, 0, 0);
#else
    *(int4*)lp = *(const int4*)gp;
#endif
  }
}

__device__ __forceinline__ void wait_async_copies() {
#if ASYNC_LDS_COPY
#  if defined(__has_builtin) && __has_builtin(__builtin_amdgcn_s_wait_asynccnt)
  __builtin_amdgcn_s_wait_asynccnt(0);
#  else
  asm volatile("s_wait_asynccnt 0" ::: "memory");
#  endif
#endif
}

// ---------------------------------------------------------------------------
// Kernel 3: fused NT-Xent — zn @ zn^T via v_wmma_f32_16x16x32_f16 with a
// fixed-shift streaming logsumexp (sim = cos/temp is bounded by 2.0, so
// lse = 2 + log(sum exp(sim-2)) is exactly stable: no running max needed).
// 4 waves/block, 16 rows/wave; B column chunks double-buffered through LDS.
// B fragments for all 4 K-chunks are preloaded (one dscnt wait per tile) and
// accumulation uses two independent WMMA chains to keep the XDL pipe busy.
// ---------------------------------------------------------------------------
__global__ void __launch_bounds__(128) ntxent_kernel(
    const _Float16* __restrict__ znh, float* __restrict__ nt_part) {
  const int tid  = threadIdx.x;
  const int lane = tid & 31;
  const int wave = tid >> 5;
  const int hi   = lane >> 4;          // lane half: selects K sub-range
  const int ln   = lane & 15;
  const int rowTile = blockIdx.x * 4 + wave;           // 0..511
  const int r0 = rowTile * 16;

  __shared__ __align__(16) _Float16 bbuf[2][CHUNK * BPAD];   // 2 x 17 KB

  // --- A fragments: this wave's 16 rows, K = 0..127 in 4 chunks of 32.
  // 16-bit A 16x32 layout: lane ln holds M=ln; VGPR v (v<4): K = 2v + 8*hi,
  // (v>=4): K = 16 + 2(v-4) + 8*hi (per chunk base 32c).
  v16h a[4];
  {
    const _Float16* arow = znh + (size_t)(r0 + ln) * ZDIM;
    #pragma unroll
    for (int c = 0; c < 4; ++c) {
      #pragma unroll
      for (int v = 0; v < 8; ++v) {
        int k = 32 * c + ((v & 4) << 2) + 8 * hi + 2 * (v & 3);
        h2 p = *(const h2*)(arow + k);
        a[c][2 * v]     = p.x;
        a[c][2 * v + 1] = p.y;
      }
    }
  }

  // Fixed-shift sumexp and positive-pair logit per lane-row (M = 8*hi + v).
  float s[8], pos[8];
  #pragma unroll
  for (int v = 0; v < 8; ++v) { s[v] = 0.f; pos[v] = 0.f; }

  stage_chunk(znh, &bbuf[0][0], tid);                  // chunk 0
  for (int cc = 0; cc < NTOT / CHUNK; ++cc) {          // 128 chunks
    const int cur = cc & 1;
    wait_async_copies();
    __syncthreads();
    if (cc + 1 < NTOT / CHUNK)
      stage_chunk(znh + (size_t)(cc + 1) * CHUNK * ZDIM, &bbuf[cur ^ 1][0], tid);

    #pragma unroll
    for (int t4 = 0; t4 < 4; ++t4) {
      const int ct = cc * 4 + t4;
      // B fragment: 32x16 16-bit layout: lane ln holds N=ln; per chunk c the
      // lane's halfs are at K = 32c + 8*hi and 32c + 16 + 8*hi (two b128 reads).
      const _Float16* bp = &bbuf[cur][(t4 * 16 + ln) * BPAD];
      v16h bf[4];
      #pragma unroll
      for (int c = 0; c < 4; ++c) {                    // preload: 8x ds_load_b128
        v8h blo = *(const v8h*)(bp + 32 * c + 8 * hi);
        v8h bhi = *(const v8h*)(bp + 32 * c + 16 + 8 * hi);
        #pragma unroll
        for (int i = 0; i < 8; ++i) { bf[c][i] = blo[i]; bf[c][8 + i] = bhi[i]; }
      }
      // Two independent accumulation chains over the 4 K-chunks.
      v8f acc0 = {}, acc1 = {};
      acc0 = __builtin_amdgcn_wmma_f32_16x16x32_f16(false, a[0], false, bf[0],
                                                    (short)0, acc0, false, false);
      acc1 = __builtin_amdgcn_wmma_f32_16x16x32_f16(false, a[1], false, bf[1],
                                                    (short)0, acc1, false, false);
      acc0 = __builtin_amdgcn_wmma_f32_16x16x32_f16(false, a[2], false, bf[2],
                                                    (short)0, acc0, false, false);
      acc1 = __builtin_amdgcn_wmma_f32_16x16x32_f16(false, a[3], false, bf[3],
                                                    (short)0, acc1, false, false);

      const int gj = ct * 16 + ln;                     // this lane's column
      #pragma unroll
      for (int v = 0; v < 8; ++v) {
        const int gi = r0 + 8 * hi + v;                // C layout: M = v + 8*hi
        float simv = (acc0[v] + acc1[v]) * INV_TEMP;
        float e = __expf(simv - SIM_MAX);              // in [e^-4, 1]
        s[v] += (gj == gi) ? 0.0f : e;                 // mask diagonal
        if (gj == (gi ^ BATCH)) pos[v] = simv;         // positive pair (i+B mod N)
      }
    }
    __syncthreads();
  }

  // Cross-lane reduction: pure sums now (xor 1,2,4,8 stay within each half).
  float lsum = 0.f;
  #pragma unroll
  for (int v = 0; v < 8; ++v) {
    float sv = s[v], pv = pos[v];
    #pragma unroll
    for (int off = 1; off <= 8; off <<= 1) {
      sv += __shfl_xor(sv, off, 32);
      pv += __shfl_xor(pv, off, 32);
    }
    lsum += (SIM_MAX + __logf(sv)) - pv;               // lse - pos for this row
  }
  lsum += __shfl_xor(lsum, 16, 32);                    // add other half's rows
  if (lane == 0) nt_part[rowTile] = lsum;
}

// ---------------------------------------------------------------------------
// Kernel 4: final deterministic reduction + loss assembly (1 block).
// ---------------------------------------------------------------------------
__global__ void finalize_kernel(const float* __restrict__ pu_part,
                                const float* __restrict__ nt_part,
                                const float* __restrict__ w_on,
                                float* __restrict__ out) {
  __shared__ float acc[9];
  const int t = threadIdx.x;
  if (t < 8) {
    float s = 0.f;
    for (int b = 0; b < 512; ++b) s += pu_part[b * 8 + t];
    acc[t] = s;
  } else if (t == 8) {
    float s = 0.f;
    for (int i = 0; i < 512; ++i) s += nt_part[i];
    acc[8] = s;
  }
  __syncthreads();
  if (t == 0) {
    float n_pos = fmaxf(1.0f, acc[0]);
    float n_unl = fmaxf(1.0f, acc[1]);
    float onnpu_sum = 0.f;
    #pragma unroll
    for (int k = 0; k < 2; ++k) {
      float ypos    = acc[2 + 3 * k];
      float yposinv = acc[3 + 3 * k];
      float yunl    = acc[4 + 3 * k];
      float prisk = PRIOR_P / n_pos * ypos;
      float nrisk = (1.0f - PRIOR_P) / (n_unl * (1.0f - PRIOR)) * yunl
                  - (1.0f - PRIOR_P) * PRIOR / (n_pos * (1.0f - PRIOR)) * yposinv;
      onnpu_sum += (nrisk < 0.0f) ? -nrisk : (prisk + nrisk);   // nnPU: beta=0, gamma=1
    }
    float onnpu_l = 0.5f * onnpu_sum;
    float nt_l = acc[8] / (float)NTOT;
    float w = w_on[0];
    out[0] = w * onnpu_l + (1.0f - w) * nt_l;
  }
}

// ---------------------------------------------------------------------------
extern "C" void kernel_launch(void* const* d_in, const int* in_sizes, int n_in,
                              void* d_out, int out_size, void* d_ws, size_t ws_size,
                              hipStream_t stream) {
  (void)in_sizes; (void)n_in; (void)out_size; (void)ws_size;
  const float* h_i    = (const float*)d_in[0];
  const float* h_j    = (const float*)d_in[1];
  const float* z_i    = (const float*)d_in[2];
  const float* z_j    = (const float*)d_in[3];
  const int*   target = (const int*)d_in[4];
  const float* W      = (const float*)d_in[5];
  const float* b      = (const float*)d_in[6];
  const float* w_on   = (const float*)d_in[7];
  float* out = (float*)d_out;

  // Workspace: [zn f16: 8192*128] [pu partials: 512*8 f32] [nt partials: 512 f32]
  _Float16* znh = (_Float16*)d_ws;
  float* pu_part = (float*)((char*)d_ws + (size_t)NTOT * ZDIM * sizeof(_Float16));
  float* nt_part = pu_part + 512 * 8;

  normalize_z_kernel<<<NTOT / 8, 256, 0, stream>>>(z_i, z_j, znh);
  gemv_pu_kernel<<<BATCH / 8, 256, 0, stream>>>(h_i, h_j, W, b, target, pu_part);
  ntxent_kernel<<<NTOT / 64, 128, 0, stream>>>(znh, nt_part);
  finalize_kernel<<<1, 32, 0, stream>>>(pu_part, nt_part, w_on, out);
}
